// FastLocalAttention_10960756540074
// MI455X (gfx1250) — compile-verified, matrix operand order checked by reading, not run on hardware
//
#include <hip/hip_runtime.h>

// ---------------------------------------------------------------------------
// FastLocalAttention on gfx1250 (MI455X): bf16 WMMA pipeline, v3
//   - global->LDS staging via GLOBAL_LOAD_ASYNC_TO_LDS_B128 (ASYNCcnt path)
//   - weights pre-transposed once -> all staging is b128-granular
//   - V produced directly in (B,H,D,S) layout by its projection GEMM
//   - 8-wave GEMM workgroups, 128x64 tile, K-step 64 (64 WMMA per barrier)
//   - attention tiles are 32-aligned => no masking needed at all
// ---------------------------------------------------------------------------

#define B_  2
#define S_  4096
#define E_  768
#define H_  12
#define D_  64
#define CCH 64
#define R_  256
#define NCHUNK 64           // S/C

typedef __attribute__((ext_vector_type(16))) __bf16 v16bf;
typedef __attribute__((ext_vector_type(8)))  float  v8f;

union Frag {
    unsigned int u[8];
    v16bf v;
};

static __device__ __forceinline__ unsigned short f2bf(float f) {
    unsigned int u = __float_as_uint(f);
    unsigned int r = u + 0x7FFFu + ((u >> 16) & 1u);   // round-to-nearest-even
    return (unsigned short)(r >> 16);
}

static __device__ __forceinline__ v8f wmma_bf16(const v16bf& a, const v16bf& b, v8f c) {
    return __builtin_amdgcn_wmma_f32_16x16x32_bf16(false, a, false, b, (short)0, c, false, false);
}

// 16-byte async global->LDS copy (no VGPR round trip, tracked by ASYNCcnt).
// LDS aperture maps addr[31:0] to the wave's LDS base, so the low 32 bits of
// a generic pointer to a __shared__ object are the LDS byte address.
static __device__ __forceinline__ void async_ld_b128(void* lds_dst, const void* gsrc) {
    unsigned int lds_off = (unsigned int)(size_t)lds_dst;
    asm volatile("global_load_async_to_lds_b128 %0, %1, off"
                 :: "v"(lds_off), "v"(gsrc)
                 : "memory");
}
static __device__ __forceinline__ void wait_async0() {
    asm volatile("s_wait_asynccnt 0x0" ::: "memory");
}

// ---------------------------------------------------------------------------
// fp32 -> bf16 converter (grid-stride)
// ---------------------------------------------------------------------------
__global__ void cvt_bf16_kernel(const float* __restrict__ src,
                                unsigned short* __restrict__ dst, long n) {
    long i = (long)blockIdx.x * blockDim.x + threadIdx.x;
    long stride = (long)gridDim.x * blockDim.x;
    for (; i < n; i += stride) dst[i] = f2bf(src[i]);
}

// fp32 W[K,N] -> bf16 Wt[N,K] (one-time transpose; writes coalesced)
__global__ void cvt_wt_kernel(const float* __restrict__ W,
                              unsigned short* __restrict__ Wt) {
    int f = blockIdx.x * blockDim.x + threadIdx.x;    // over N*K = 589824
    if (f >= E_ * E_) return;
    int n = f / E_, k = f - n * E_;
    Wt[f] = f2bf(W[(size_t)k * E_ + n]);
}

// ---------------------------------------------------------------------------
// bf16 GEMM:  C[M,N] = A[M,K] @ Wt[N,K]^T + bias
//   256 threads (8 waves), tile 128(M) x 64(N), K-step 64
//   mode 0: bf16((acc+bias)*scale) -> (B,H,S,D)
//   mode 2: bf16((acc+bias))       -> (B,H,D,S)   (transposed V)
//   mode 1: fp32 (acc+bias)        -> row-major (B*S, E)
// ---------------------------------------------------------------------------
__global__ __launch_bounds__(256) void gemm_bf16_kernel(
    const unsigned short* __restrict__ A,    // M x K row-major
    const unsigned short* __restrict__ Wt,   // N x K row-major (pre-transposed)
    const float* __restrict__ bias,
    float* __restrict__ out_f32,
    unsigned short* __restrict__ out_sc,
    int M, int N, int K, float scale, int mode) {

    __shared__ unsigned short Alds[128 * 64];   // (m, k)
    __shared__ unsigned short Btlds[64 * 64];   // (n, k)

    const int tid  = threadIdx.x;
    const int lane = tid & 31;
    const int wave = tid >> 5;
    const int g    = lane >> 4;
    const int l16  = lane & 15;
    const int tile_n = blockIdx.x * 64;
    const int tile_m = blockIdx.y * 128;

    v8f acc[4];
    for (int j = 0; j < 4; j++)
        for (int r = 0; r < 8; r++) acc[j][r] = 0.f;

    for (int k0 = 0; k0 < K; k0 += 64) {
        // branchless prefetch distance (0 on last step keeps address in range)
        const int pfd = (k0 + 64 < K) ? 64 : 0;

        // --- stage A tile 128x64: 1024 async b128, 4 per thread ---
        #pragma unroll
        for (int i = 0; i < 4; i++) {
            int idx = tid + i * 256;
            int r = idx >> 3, c4 = idx & 7;
            const unsigned short* src = A + (size_t)(tile_m + r) * K + k0 + c4 * 8;
            async_ld_b128(Alds + r * 64 + c4 * 8, src);
            __builtin_prefetch(src + pfd, 0, 3);   // global_prefetch_b8
        }
        // --- stage Wt tile 64x64: 512 async b128, 2 per thread ---
        #pragma unroll
        for (int i = 0; i < 2; i++) {
            int idx = tid + i * 256;
            int r = idx >> 3, c4 = idx & 7;
            const unsigned short* src = Wt + (size_t)(tile_n + r) * K + k0 + c4 * 8;
            async_ld_b128(Btlds + r * 64 + c4 * 8, src);
        }
        wait_async0();
        __syncthreads();

        #pragma unroll
        for (int kt = 0; kt < 2; kt++) {
            Frag af;
            *(uint4*)&af.u[0] = *(const uint4*)(Alds + (wave * 16 + l16) * 64 + kt * 32 + g * 8);
            *(uint4*)&af.u[4] = *(const uint4*)(Alds + (wave * 16 + l16) * 64 + kt * 32 + 16 + g * 8);
            #pragma unroll
            for (int nt = 0; nt < 4; nt++) {
                Frag bf;
                *(uint4*)&bf.u[0] = *(const uint4*)(Btlds + (nt * 16 + l16) * 64 + kt * 32 + g * 8);
                *(uint4*)&bf.u[4] = *(const uint4*)(Btlds + (nt * 16 + l16) * 64 + kt * 32 + 16 + g * 8);
                acc[nt] = wmma_bf16(af.v, bf.v, acc[nt]);
            }
        }
        __syncthreads();
    }

    // --- epilogue: C element (M = r + 8g, N = l16) lives in acc[nt][r] ---
    #pragma unroll
    for (int nt = 0; nt < 4; nt++) {
        int ncol = tile_n + nt * 16 + l16;
        float bv = bias[ncol];
        int h = ncol >> 6;             // / D_
        int d = ncol & (D_ - 1);
        #pragma unroll
        for (int r = 0; r < 8; r++) {
            int grow = tile_m + wave * 16 + r + 8 * g;   // global row in [0, B*S)
            float v = acc[nt][r] + bv;
            int b = grow >> 12;            // / S_
            int s = grow & (S_ - 1);
            if (mode == 0) {
                out_sc[((((size_t)b * H_ + h) * S_) + s) * D_ + d] = f2bf(v * scale);
            } else if (mode == 2) {
                out_sc[((((size_t)b * H_ + h) * D_) + d) * S_ + s] = f2bf(v);
            } else {
                out_f32[(size_t)grow * N + ncol] = v;
            }
        }
    }
}

// ---------------------------------------------------------------------------
// Local attention, flash-style online softmax.
//   grid = B*H*nC blocks of 128 threads (4 waves); wave w owns q-rows 16w..16w+15
//   window walked in 32-key tiles; tiles are 32-aligned => fully valid or skipped
// ---------------------------------------------------------------------------
__global__ __launch_bounds__(128) void local_attn_kernel(
    const unsigned short* __restrict__ Q,    // (B,H,S,D) bf16, pre-scaled by 1/sqrt(D)
    const unsigned short* __restrict__ Kb,   // (B,H,S,D) bf16
    const unsigned short* __restrict__ Vt,   // (B,H,D,S) bf16 (transposed)
    unsigned short* __restrict__ ctx)        // (B,S,E)   bf16
{
    __shared__ unsigned short Klds[32 * 64];     // (key, d)
    __shared__ unsigned short Vtlds[64 * 32];    // (d, key)
    __shared__ unsigned short Plds[4][16 * 32];  // per-wave prob tile (qrow, key)

    const int blk   = blockIdx.x;
    const int chunk = blk & (NCHUNK - 1);
    const int bh    = blk >> 6;                  // b*H + h
    const int h     = bh % H_;
    const int b     = bh / H_;
    const int tid   = threadIdx.x;
    const int lane  = tid & 31;
    const int wave  = tid >> 5;
    const int g     = lane >> 4;
    const int l16   = lane & 15;

    const unsigned short* qbase = Q  + ((size_t)bh * S_ + chunk * CCH) * D_;
    const unsigned short* kbp   = Kb + (size_t)bh * S_ * D_;
    const unsigned short* vtp   = Vt + (size_t)bh * D_ * S_;

    // Q fragments (rows wave*16..+15, d split into two 32-wide K-tiles)
    Frag qf[2];
    #pragma unroll
    for (int dt = 0; dt < 2; dt++) {
        *(uint4*)&qf[dt].u[0] = *(const uint4*)(qbase + (wave * 16 + l16) * D_ + dt * 32 + g * 8);
        *(uint4*)&qf[dt].u[4] = *(const uint4*)(qbase + (wave * 16 + l16) * D_ + dt * 32 + 16 + g * 8);
    }

    v8f acc[4];
    float mrow[8], lrow[8];
    #pragma unroll
    for (int r = 0; r < 8; r++) { mrow[r] = -__builtin_inff(); lrow[r] = 0.f; }
    for (int j = 0; j < 4; j++)
        for (int r = 0; r < 8; r++) acc[j][r] = 0.f;

    // valid tile range: base = chunk*64 - 256 + 32t must lie in [0, S-32]
    const int tstart = (8 - 2 * chunk) > 0 ? (8 - 2 * chunk) : 0;
    const int tend   = (136 - 2 * chunk) < 18 ? (136 - 2 * chunk) : 18;

    for (int t = tstart; t < tend; t++) {
        const int base_pos = chunk * CCH - R_ + t * 32;   // always in [0, S-32]

        // --- stage K tile (32 keys x 64 d): 256 async b128, 2 per thread ---
        #pragma unroll
        for (int i = 0; i < 2; i++) {
            int idx = tid + i * 128;
            int kr = idx >> 3, c4 = idx & 7;
            async_ld_b128(Klds + kr * 64 + c4 * 8,
                          kbp + (size_t)(base_pos + kr) * D_ + c4 * 8);
        }
        // --- stage Vt tile (64 d x 32 keys): 256 async b128, 2 per thread ---
        #pragma unroll
        for (int i = 0; i < 2; i++) {
            int idx = tid + i * 128;
            int dr = idx >> 2, c4 = idx & 3;
            async_ld_b128(Vtlds + dr * 32 + c4 * 8,
                          vtp + (size_t)dr * S_ + base_pos + c4 * 8);
        }
        wait_async0();
        __syncthreads();

        // --- scores: two 16x16 C-tiles (keys 0..15 and 16..31) ---
        v8f sc[2];
        #pragma unroll
        for (int n0 = 0; n0 < 2; n0++) {
            v8f s;
            for (int r = 0; r < 8; r++) s[r] = 0.f;
            #pragma unroll
            for (int dt = 0; dt < 2; dt++) {
                Frag kf;  // B frag: element (dd, key) = K[key][dd], contiguous in d
                *(uint4*)&kf.u[0] = *(const uint4*)(Klds + (n0 * 16 + l16) * 64 + dt * 32 + g * 8);
                *(uint4*)&kf.u[4] = *(const uint4*)(Klds + (n0 * 16 + l16) * 64 + dt * 32 + 16 + g * 8);
                s = wmma_bf16(qf[dt].v, kf.v, s);
            }
            sc[n0] = s;
        }

        // --- online softmax: row stats via shfl-xor within 16-lane halves ---
        float alpha[8];
        #pragma unroll
        for (int r = 0; r < 8; r++) {
            float v = fmaxf(sc[0][r], sc[1][r]);
            for (int off = 1; off < 16; off <<= 1)
                v = fmaxf(v, __shfl_xor(v, off, 32));
            float mnew = fmaxf(mrow[r], v);
            alpha[r] = __expf(mrow[r] - mnew);
            mrow[r] = mnew;
        }
        #pragma unroll
        for (int r = 0; r < 8; r++) {
            float p0 = __expf(sc[0][r] - mrow[r]);
            float p1 = __expf(sc[1][r] - mrow[r]);
            Plds[wave][(r + 8 * g) * 32 + l16]      = f2bf(p0);
            Plds[wave][(r + 8 * g) * 32 + 16 + l16] = f2bf(p1);
            float v = p0 + p1;
            for (int off = 1; off < 16; off <<= 1)
                v += __shfl_xor(v, off, 32);
            lrow[r] = lrow[r] * alpha[r] + v;
        }
        #pragma unroll
        for (int j = 0; j < 4; j++)
            for (int r = 0; r < 8; r++) acc[j][r] *= alpha[r];
        __syncthreads();

        // --- acc += P @ V  (P as A-fragment via LDS transpose pass) ---
        Frag pf;
        *(uint4*)&pf.u[0] = *(const uint4*)(&Plds[wave][l16 * 32 + g * 8]);
        *(uint4*)&pf.u[4] = *(const uint4*)(&Plds[wave][l16 * 32 + 16 + g * 8]);
        #pragma unroll
        for (int dt = 0; dt < 4; dt++) {
            Frag vf;  // B frag: element (key, dcol) = Vt[d][key], contiguous in key
            *(uint4*)&vf.u[0] = *(const uint4*)(Vtlds + (dt * 16 + l16) * 32 + g * 8);
            *(uint4*)&vf.u[4] = *(const uint4*)(Vtlds + (dt * 16 + l16) * 32 + 16 + g * 8);
            acc[dt] = wmma_bf16(pf.v, vf.v, acc[dt]);
        }
        __syncthreads();
    }

    // --- normalize and write ctx (B,S,E) bf16 ---
    unsigned short* cb = ctx + ((size_t)b * S_ + chunk * CCH) * E_ + h * D_;
    #pragma unroll
    for (int dt = 0; dt < 4; dt++) {
        #pragma unroll
        for (int r = 0; r < 8; r++) {
            int srow = wave * 16 + r + 8 * g;
            float v = acc[dt][r] / lrow[r];
            cb[(size_t)srow * E_ + dt * 16 + l16] = f2bf(v);
        }
    }
}

// ---------------------------------------------------------------------------
extern "C" void kernel_launch(void* const* d_in, const int* in_sizes, int n_in,
                              void* d_out, int out_size, void* d_ws, size_t ws_size,
                              hipStream_t stream) {
    const float* hs = (const float*)d_in[0];
    const float* Wq = (const float*)d_in[1];
    const float* bq = (const float*)d_in[2];
    const float* Wk = (const float*)d_in[3];
    const float* bk = (const float*)d_in[4];
    const float* Wv = (const float*)d_in[5];
    const float* bv = (const float*)d_in[6];
    const float* Wo = (const float*)d_in[7];
    const float* bo = (const float*)d_in[8];
    float* out = (float*)d_out;

    const size_t nHS = (size_t)B_ * S_ * E_;   // 6,291,456
    const size_t nW  = (size_t)E_ * E_;        // 589,824

    char* ws = (char*)d_ws;
    size_t off = 0;
    unsigned short* hsb  = (unsigned short*)(ws + off); off += nHS * 2;
    unsigned short* wqt  = (unsigned short*)(ws + off); off += nW * 2;   // transposed
    unsigned short* wkt  = (unsigned short*)(ws + off); off += nW * 2;
    unsigned short* wvt  = (unsigned short*)(ws + off); off += nW * 2;
    unsigned short* wot  = (unsigned short*)(ws + off); off += nW * 2;
    unsigned short* qb   = (unsigned short*)(ws + off); off += nHS * 2;  // (B,H,S,D)
    unsigned short* kb2  = (unsigned short*)(ws + off); off += nHS * 2;  // (B,H,S,D)
    unsigned short* vtb  = (unsigned short*)(ws + off); off += nHS * 2;  // (B,H,D,S)
    unsigned short* ctxb = (unsigned short*)(ws + off); off += nHS * 2;  // (B,S,E)

    cvt_bf16_kernel<<<2048, 256, 0, stream>>>(hs, hsb, (long)nHS);
    cvt_wt_kernel<<<(int)((nW + 255) / 256), 256, 0, stream>>>(Wq, wqt);
    cvt_wt_kernel<<<(int)((nW + 255) / 256), 256, 0, stream>>>(Wk, wkt);
    cvt_wt_kernel<<<(int)((nW + 255) / 256), 256, 0, stream>>>(Wv, wvt);
    cvt_wt_kernel<<<(int)((nW + 255) / 256), 256, 0, stream>>>(Wo, wot);

    const int M = B_ * S_;
    dim3 gg(E_ / 64, M / 128);  // (12, 64)
    gemm_bf16_kernel<<<gg, 256, 0, stream>>>(hsb, wqt, bq, nullptr, qb,  M, E_, E_, 0.125f, 0);
    gemm_bf16_kernel<<<gg, 256, 0, stream>>>(hsb, wkt, bk, nullptr, kb2, M, E_, E_, 1.0f,   0);
    gemm_bf16_kernel<<<gg, 256, 0, stream>>>(hsb, wvt, bv, nullptr, vtb, M, E_, E_, 1.0f,   2);

    local_attn_kernel<<<B_ * H_ * NCHUNK, 128, 0, stream>>>(qb, kb2, vtb, ctxb);

    // output projection: fp32 result
    gemm_bf16_kernel<<<gg, 256, 0, stream>>>(ctxb, wot, bo, out, nullptr, M, E_, E_, 1.0f, 1);
}
